// ReIdHead_42812234006933
// MI455X (gfx1250) — matrix operand order repaired and checked by reading, not compiled, not run on hardware
//
#include <hip/hip_runtime.h>
#include <hip/hip_bf16.h>
#include <stdint.h>

#define B_Q   256
#define DIM   256
#define N_DB  500000
#define NTILES (N_DB / 16)     // 31250 exactly
#define EPS   1e-8f

typedef __bf16 bf16_t;
typedef bf16_t v16bf __attribute__((ext_vector_type(16)));
typedef bf16_t v8bf  __attribute__((ext_vector_type(8)));
typedef float  v8f   __attribute__((ext_vector_type(8)));
typedef float  v4f   __attribute__((ext_vector_type(4)));

// Padded LDS row: 256 bf16 + 8 pad = 264 elements = 528 bytes.
// Row-to-row bank shift = 132 dwords % 64 = 4 banks -> 16 rows hit 16 distinct banks.
#define ROWB 528
#define Q_LDS_BYTES (B_Q * ROWB)                  // 135168
#define DB_LDS_OFF  Q_LDS_BYTES
#define DB_LDS_BYTES (16 * ROWB)                  // 8448 per buffer, x2 (double buffered)
#define SCALE_OFF   (DB_LDS_OFF + 2 * DB_LDS_BYTES)
#define SMEM_BYTES  (SCALE_OFF + 128)             // 152192 B < 320 KB WGP LDS (2 blocks fit)

#if defined(__has_builtin)
#if __has_builtin(__builtin_amdgcn_sched_group_barrier)
#define SCHED_GROUP(mask, size, id) __builtin_amdgcn_sched_group_barrier(mask, size, id)
#endif
#endif
#ifndef SCHED_GROUP
#define SCHED_GROUP(mask, size, id)
#endif

__device__ __forceinline__ unsigned f32_to_bf16u(float f) {
    unsigned u = __float_as_uint(f);                 // round-to-nearest-even
    return (u + 0x7FFFu + ((u >> 16) & 1u)) >> 16;
}
__device__ __forceinline__ unsigned pack_bf16(float lo, float hi) {
    return f32_to_bf16u(lo) | (f32_to_bf16u(hi) << 16);
}
// monotonic float -> u32 map (order-preserving as unsigned)
__device__ __forceinline__ unsigned fmap(float f) {
    unsigned u = __float_as_uint(f);
    return u ^ ((u >> 31) ? 0xFFFFFFFFu : 0x80000000u);
}
__device__ __forceinline__ float finvmap(unsigned u) {
    unsigned b = (u & 0x80000000u) ? (u ^ 0x80000000u) : ~u;
    return __uint_as_float(b);
}
// streamed-once database reads: non-temporal so they don't evict L2-resident queries
__device__ __forceinline__ v4f ntload4(const float* p) {
    return __builtin_nontemporal_load((const v4f*)p);
}

// row-norm (exact fp32, wave32 shuffle reduce) + bf16 pack + LDS stage of one db row
__device__ __forceinline__ void stage_tile(v4f fa, v4f fb, char* dbbuf, float* rs,
                                           int w, int lane) {
    float ss = fa.x*fa.x + fa.y*fa.y + fa.z*fa.z + fa.w*fa.w
             + fb.x*fb.x + fb.y*fb.y + fb.z*fb.z + fb.w*fb.w;
    #pragma unroll
    for (int mk = 16; mk >= 1; mk >>= 1) ss += __shfl_xor(ss, mk, 32);
    if (lane == 0) rs[w] = 1.0f / fmaxf(sqrtf(ss), EPS);
    uint4 pk;
    pk.x = pack_bf16(fa.x, fa.y); pk.y = pack_bf16(fa.z, fa.w);
    pk.z = pack_bf16(fb.x, fb.y); pk.w = pack_bf16(fb.z, fb.w);
    *(uint4*)(dbbuf + w * ROWB + lane * 16) = pk;
}

// ---------------- prep: normalize queries (fp32 math) -> bf16 q_n in ws; zero keys ----------------
__global__ __launch_bounds__(256) void prep_kernel(const float* __restrict__ q,
                                                   unsigned short* __restrict__ qn,
                                                   unsigned long long* __restrict__ keys) {
    __shared__ float red[256];
    const int r = blockIdx.x, t = threadIdx.x;
    const float v = q[r * DIM + t];
    red[t] = v * v;
    __syncthreads();
    for (int s = 128; s > 0; s >>= 1) { if (t < s) red[t] += red[t + s]; __syncthreads(); }
    const float scale = 1.0f / fmaxf(sqrtf(red[0]), EPS);
    qn[r * DIM + t] = (unsigned short)f32_to_bf16u(v * scale);
    if (t == 0) keys[r] = 0ull;   // any real packed key > 0
}

// ---------------- main: stream db tiles, bf16 WMMA, running argmax ----------------
__global__ __launch_bounds__(512) void knn_kernel(const unsigned short* __restrict__ qn,
                                                  const float* __restrict__ db,
                                                  unsigned long long* __restrict__ keys) {
    extern __shared__ char smem[];
    char*  qld      = smem;
    char*  dbld     = smem + DB_LDS_OFF;           // two 16-row bf16 buffers
    float* rowscale = (float*)(smem + SCALE_OFF);  // two 16-entry buffers

    const int tid  = threadIdx.x;
    const int w    = tid >> 5;     // wave id 0..15 (owns query tile w; loads db row w)
    const int lane = tid & 31;
    const int m    = lane & 15;
    const int half = lane >> 4;

    // ---- stage all 256 normalized bf16 query rows into LDS (L2-resident source) ----
    {
        const int r = tid >> 1, h = tid & 1;
        const uint4* src = (const uint4*)(qn + r * DIM + h * 128);   // 256 B per thread
        uint4*       dst = (uint4*)(qld + r * ROWB + h * 256);
        #pragma unroll
        for (int i = 0; i < 16; ++i) dst[i] = src[i];
    }
    __syncthreads();

    // ---- cache this wave's A operand (16 queries x 256 K, bf16) in 64 VGPRs ----
    // A 16x32 bf16 layout: lanes 0-15: V0-3=K[0..7], V4-7=K[16..23]; lanes 16-31: +8 / +24.
    v16bf afrag[8];
    {
        const char* qrow = qld + (w * 16 + m) * ROWB;
        #pragma unroll
        for (int c = 0; c < 8; ++c) {
            v8bf a0 = *(const v8bf*)(qrow + c * 64 + half * 16);
            v8bf a1 = *(const v8bf*)(qrow + c * 64 + half * 16 + 32);
            afrag[c] = __builtin_shufflevector(a0, a1,
                        0,1,2,3,4,5,6,7,8,9,10,11,12,13,14,15);
        }
    }

    float rm[8]; int ri[8];
    #pragma unroll
    for (int v = 0; v < 8; ++v) { rm[v] = -1e30f; ri[v] = 0; }

    const int stride = (int)gridDim.x;
    const int t0     = (int)blockIdx.x;   // t0 < NTILES always (grid 1250 << 31250)

    // prologue: stage tile t0 into buffer 0
    {
        const float* p = db + ((size_t)(t0 * 16 + w)) * DIM + lane * 8;
        v4f fa = ntload4(p), fb = ntload4(p + 4);
        stage_tile(fa, fb, dbld, rowscale, w, lane);
    }

    int i = 0;
    for (int t = t0; t < NTILES; t += stride, ++i) {
        const int  buf  = i & 1;
        const int  tn   = t + stride;
        const bool more = tn < NTILES;

        __syncthreads();   // buf now holds tile t; no global loads outstanding here

        // issue next tile's HBM loads AFTER the barrier so __syncthreads()'s
        // conservative loadcnt wait can't serialize them; they overlap the
        // whole WMMA + argmax phase and are first waited on in stage_tile().
        v4f na = {0.f,0.f,0.f,0.f}, nb = {0.f,0.f,0.f,0.f};
        if (more) {
            const float* p = db + ((size_t)(tn * 16 + w)) * DIM + lane * 8;
            na = ntload4(p); nb = ntload4(p + 4);
        }

        // B fragments for 8 K-chunks.
        // B 32x16 bf16 layout: lane n<16: V0-3=K[0..7], V4-7=K[16..23]; lane n+16: +8 / +24
        const char* drow = dbld + buf * DB_LDS_BYTES + m * ROWB + half * 16;
        v16bf bfrag[8];
        #pragma unroll
        for (int c = 0; c < 8; ++c) {
            v8bf b0 = *(const v8bf*)(drow + c * 64);
            v8bf b1 = *(const v8bf*)(drow + c * 64 + 32);
            bfrag[c] = __builtin_shufflevector(b0, b1,
                        0,1,2,3,4,5,6,7,8,9,10,11,12,13,14,15);
        }
        v8f acc = {0.f,0.f,0.f,0.f,0.f,0.f,0.f,0.f};
        #pragma unroll
        for (int c = 0; c < 8; ++c)
            acc = __builtin_amdgcn_wmma_f32_16x16x32_bf16(false, afrag[c], false, bfrag[c],
                                                          (short)0, acc, false, false);

        // scale by 1/||d_row|| and update running argmax (lane column m = db row t*16+m)
        const float s  = rowscale[buf * 16 + m];
        const int  idx = t * 16 + m;
        #pragma unroll
        for (int v = 0; v < 8; ++v) {
            const float val = acc[v] * s;
            if (val > rm[v]) { rm[v] = val; ri[v] = idx; }
        }

        // Pin the LDS-load / WMMA software pipeline (mask 0x8 = MFMA/WMMA,
        // 0x100 = DS read): frag c+1's loads are forced above WMMA c, so the
        // fragments must live in distinct registers and each WMMA waits only
        // on loads issued one WMMA earlier instead of s_wait_dscnt 0.
        SCHED_GROUP(0x100, 4, 0);                 // loads: frag 0, frag 1
        #pragma unroll
        for (int c = 0; c < 6; ++c) {
            SCHED_GROUP(0x008, 1, 0);             // WMMA c
            SCHED_GROUP(0x100, 2, 0);             // loads: frag c+2
        }
        SCHED_GROUP(0x008, 2, 0);                 // WMMA 6, 7
        SCHED_GROUP(0x100, 1, 0);                 // rowscale b32 load

        // stage tile t+stride into the other buffer (its readers finished before this
        // iteration's barrier) — single barrier per tile
        if (more)
            stage_tile(na, nb, dbld + (buf ^ 1) * DB_LDS_BYTES,
                       rowscale + (buf ^ 1) * 16, w, lane);
    }

    // ---- block-local reduction over the 16 lane-columns (overlay on query LDS) ----
    float* redv = (float*)qld;             // [256][16]
    int*   redi = (int*)(qld + 16384);     // [256][16]
    #pragma unroll
    for (int v = 0; v < 8; ++v) {
        const int q = w * 16 + v + half * 8;   // D-matrix row for VGPR v / half-wave
        redv[q * 16 + m] = rm[v];
        redi[q * 16 + m] = ri[v];
    }
    __syncthreads();
    if (tid < 256) {
        float best = -1e30f; int bi = 0;
        #pragma unroll
        for (int c = 0; c < 16; ++c) {
            const float v = redv[tid * 16 + c];
            const int   i = redi[tid * 16 + c];
            if (v > best) { best = v; bi = i; }
        }
        const unsigned long long key = ((unsigned long long)fmap(best) << 32) | (unsigned)bi;
        atomicMax(&keys[tid], key);    // global_atomic_max_u64 merge across blocks
    }
}

// ---------------- finalize: unpack keys -> top_sim, gather labels ----------------
__global__ __launch_bounds__(256) void final_kernel(const unsigned long long* __restrict__ keys,
                                                    const int* __restrict__ labels,
                                                    float* __restrict__ out) {
    const int t = threadIdx.x;
    const unsigned long long k = keys[t];
    const unsigned idx = (unsigned)(k & 0xFFFFFFFFull);
    out[t]        = finvmap((unsigned)(k >> 32));
    out[B_Q + t]  = (float)labels[idx];
}

extern "C" void kernel_launch(void* const* d_in, const int* in_sizes, int n_in,
                              void* d_out, int out_size, void* d_ws, size_t ws_size,
                              hipStream_t stream) {
    (void)in_sizes; (void)n_in; (void)out_size; (void)ws_size;
    const float* q      = (const float*)d_in[0];
    const float* db     = (const float*)d_in[1];
    const int*   labels = (const int*)d_in[2];

    unsigned long long* keys = (unsigned long long*)d_ws;                   // 256 * 8 B
    unsigned short*     qn   = (unsigned short*)((char*)d_ws + 2048);       // 256*256 bf16

    prep_kernel<<<B_Q, 256, 0, stream>>>(q, qn, keys);
    knn_kernel<<<1250, 512, SMEM_BYTES, stream>>>(qn, db, keys);
    final_kernel<<<1, 256, 0, stream>>>(keys, labels, (float*)d_out);
}
